// MLPP_1262720385099
// MI455X (gfx1250) — compile-verified, matrix operand order checked by reading, not run on hardware
//
#include <hip/hip_runtime.h>
#include <hip/hip_bf16.h>

// ---------------------------------------------------------------------------
// MI455X (gfx1250): all 8 linears of the model are 256->256 GEMMs over 131072
// rows, differing only in the gather/scatter of the contracted axis:
//   off(k) = (k>>4)*S1 + (k&15)*S0
// One templated bf16-WMMA GEMM kernel (v_wmma_f32_16x16x32_bf16, wave32)
// covers all stages. Elementwise work (the (1+att)*(h+w+c) mix and the three
// residual adds) is fused into gathers/epilogues, so the whole model is
// 8 GEMM launches + one-time weight swizzle.
//
// Block: 256 threads = 8 waves. Output tile 64(M) x 256(N) per block:
//   - A tile (64x256) gathered -> bf16 -> LDS in WMMA-fragment layout (32 KB)
//   - each wave owns 2 N-tiles and all 4 M-tiles -> B fragments are loaded
//     from L2 exactly once per block (128 KB, zero redundancy) and reused
//     across 4 WMMAs each.
// ---------------------------------------------------------------------------

typedef __attribute__((ext_vector_type(8)))  float  v8f;
typedef __attribute__((ext_vector_type(16))) __bf16 v16bf;

#define NTOT   33554432   // 4*32*32*32*256 elements
#define NROWS  131072     // NTOT / 256
#define MODE_CONT 0
#define MODE_ATT  1
#define MODE_H    2
#define MODE_W    3
#define MODE_D    4
#define EPI_STORE 0
#define EPI_ACC   1
#define EPI_RES   2

static __device__ __forceinline__ unsigned short f32_to_bf16(float f) {
  unsigned int u = __float_as_uint(f);
  unsigned int r = u + 0x7FFFu + ((u >> 16) & 1u);   // round-to-nearest-even
  return (unsigned short)(r >> 16);
}

// Base element offset of logical row r (the non-contracted indices), per stage.
template <int MODE>
static __device__ __forceinline__ int rowBase(int r) {
  if constexpr (MODE == MODE_CONT) {
    return r << 8;
  } else if constexpr (MODE == MODE_ATT) {
    int c  = r & 255, d = (r >> 8) & 31, w1 = (r >> 13) & 1,
        h1 = (r >> 14) & 1, b = r >> 15;
    return (b << 23) | (h1 << 22) | (w1 << 17) | (d << 8) | c;
  } else if constexpr (MODE == MODE_H) {
    int d = r & 31, lc = (r >> 5) & 15, w = (r >> 9) & 31,
        h1 = (r >> 14) & 1, b = r >> 15;
    return (b << 23) | (h1 << 22) | (w << 13) | (d << 8) | (lc << 4);
  } else if constexpr (MODE == MODE_W) {
    int d = r & 31, lc = (r >> 5) & 15, w1 = (r >> 9) & 1,
        h = (r >> 10) & 31, b = r >> 15;
    return (b << 23) | (h << 18) | (w1 << 17) | (d << 8) | (lc << 4);
  } else { // MODE_D
    int l = r & 15, d1 = (r >> 4) & 1, w = (r >> 5) & 31,
        h = (r >> 10) & 31, b = r >> 15;
    return (b << 23) | (h << 18) | (w << 13) | (d1 << 12) | (l << 4);
  }
}

// Element offset of contracted/produced index k relative to rowBase.
template <int MODE>
static __device__ __forceinline__ int koff(int k) {
  if constexpr (MODE == MODE_CONT) return k;
  else if constexpr (MODE == MODE_ATT) return ((k >> 4) << 18) | ((k & 15) << 13);
  else if constexpr (MODE == MODE_H)   return ((k >> 4) << 18) | (k & 15);
  else if constexpr (MODE == MODE_W)   return ((k >> 4) << 13) | (k & 15);
  else                                 return ((k >> 4) << 8)  | (k & 15); // D
}

// 16-bit WMMA operand fragment position (ISA 7.12.2): lane = m + h*16,
// element j with K(j,h) = (j&7) + (j>>3)*16 + h*8.
static __device__ __forceinline__ int frag_j(int k) { return (k & 7) | ((k >> 4) << 3); }
static __device__ __forceinline__ int frag_h(int k) { return (k >> 3) & 1; }

// ---------------------------------------------------------------------------
// Weight pre-swizzle: W (256x256 f32 [k][n]) -> bf16 B-fragments so a wave
// fetches each v16bf B operand as one contiguous 32B load per lane.
// ---------------------------------------------------------------------------
__global__ __launch_bounds__(256) void swizzle_w_kernel(
    const float* __restrict__ W, unsigned short* __restrict__ dst) {
  int t = blockIdx.x * 256 + threadIdx.x;        // 0..32767
  int jp    = t & 7;                             // j = jp*2
  int lane  = (t >> 3) & 31;
  int kstep = (t >> 8) & 7;
  int ntile = t >> 11;
  int n = ntile * 16 + (lane & 15);
  int h = lane >> 4;
  int j0 = jp * 2;
  int k0 = kstep * 32 + (j0 & 7) + ((j0 >> 3) << 4) + h * 8;
  unsigned int packed = (unsigned int)f32_to_bf16(W[k0 * 256 + n]) |
                        ((unsigned int)f32_to_bf16(W[(k0 + 1) * 256 + n]) << 16);
  *(unsigned int*)&dst[(ntile * 8 + kstep) * 512 + lane * 16 + j0] = packed;
}

// ---------------------------------------------------------------------------
// Permuted GEMM stage.
//   MIX:  A element = (1 + X[a]) * X2[a]   (fuses the attention mix)
//   EPI_STORE: Y[addr] = acc + bias
//   EPI_ACC:   Y[addr] += acc + bias       (builds h+w+c sum in place)
//   EPI_RES:   Y[addr] = R[addr] + acc + bias  (fuses residual adds)
// ---------------------------------------------------------------------------
template <int MODE, int EPI, bool MIX>
__global__ __launch_bounds__(256) void gemm_stage_kernel(
    const float* __restrict__ X, const float* __restrict__ X2,
    const unsigned short* __restrict__ Wb, const float* __restrict__ bias,
    const float* __restrict__ R, float* __restrict__ Y) {
  __shared__ __align__(32) unsigned short ldsA[64 * 256];  // 32 KB, frag layout

  const int tid = threadIdx.x;
  const int rowBlock = blockIdx.x * 64;

  // Cooperative gather: 64 rows x 256 K = 8192 k-pairs over 256 threads.
  #pragma unroll
  for (int i = 0; i < 32; ++i) {
    int p = tid + i * 256;
    int m = p >> 7;                   // 0..63
    int k = (p & 127) << 1;           // even k
    int base = rowBase<MODE>(rowBlock + m);
    int a0 = base + koff<MODE>(k);
    int a1 = base + koff<MODE>(k + 1);
    float f0, f1;
    if constexpr (MIX) {
      f0 = (1.f + X[a0]) * X2[a0];
      f1 = (1.f + X[a1]) * X2[a1];
    } else {
      f0 = X[a0];
      f1 = X[a1];
    }
    unsigned int packed = (unsigned int)f32_to_bf16(f0) |
                          ((unsigned int)f32_to_bf16(f1) << 16);
    int mtile = m >> 4, kstep = k >> 5, kk = k & 31;
    int j = frag_j(kk);               // even
    int lane = (m & 15) + (frag_h(kk) << 4);
    *(unsigned int*)&ldsA[(mtile * 8 + kstep) * 512 + lane * 16 + j] = packed;
  }
  __syncthreads();

  const int wave = tid >> 5;          // owns ntiles {2*wave, 2*wave+1}
  const int lane = tid & 31;

  v8f acc[4][2] = {};                 // [mtile][ntile] f32 accumulators

  #pragma unroll
  for (int ks = 0; ks < 8; ++ks) {
    v16bf a[4];
    #pragma unroll
    for (int mt = 0; mt < 4; ++mt)
      a[mt] = *(const v16bf*)&ldsA[(mt * 8 + ks) * 512 + lane * 16];
    #pragma unroll
    for (int nt = 0; nt < 2; ++nt) {
      int ntile = wave * 2 + nt;
      v16bf b = *(const v16bf*)&Wb[(ntile * 8 + ks) * 512 + lane * 16];
      #pragma unroll
      for (int mt = 0; mt < 4; ++mt)
        acc[mt][nt] = __builtin_amdgcn_wmma_f32_16x16x32_bf16(
            false, a[mt], false, b, (short)0, acc[mt][nt], false, false);
    }
  }

  // Epilogue: C/D layout -> lane = n (0..15 twice), VGPR r -> m = r + (lane>=16)*8
  const int nlocal = lane & 15;
  const int halfl = lane >> 4;
  #pragma unroll
  for (int nt = 0; nt < 2; ++nt) {
    int n = (wave * 2 + nt) * 16 + nlocal;
    float bv = bias[n];
    int noff = koff<MODE>(n);
    #pragma unroll
    for (int mt = 0; mt < 4; ++mt) {
      #pragma unroll
      for (int r = 0; r < 8; ++r) {
        int m = mt * 16 + halfl * 8 + r;
        int addr = rowBase<MODE>(rowBlock + m) + noff;
        float v = acc[mt][nt][r] + bv;
        if constexpr (EPI == EPI_ACC)      Y[addr] += v;
        else if constexpr (EPI == EPI_RES) Y[addr] = R[addr] + v;
        else                               Y[addr] = v;
      }
    }
  }
}

// ---------------------------------------------------------------------------
extern "C" void kernel_launch(void* const* d_in, const int* in_sizes, int n_in,
                              void* d_out, int out_size, void* d_ws, size_t ws_size,
                              hipStream_t stream) {
  (void)in_sizes; (void)n_in; (void)out_size; (void)ws_size;
  // setup_inputs order: x, W_h, b_h, W_w, b_w, W_c, b_c, W_d, b_d,
  //                     W_attn, b_attn, W_1, b_1, W_2, b_2, W_3, b_3
  const float* x      = (const float*)d_in[0];
  const float* W_h    = (const float*)d_in[1];
  const float* b_h    = (const float*)d_in[2];
  const float* W_w    = (const float*)d_in[3];
  const float* b_w    = (const float*)d_in[4];
  const float* W_c    = (const float*)d_in[5];
  const float* b_c    = (const float*)d_in[6];
  const float* W_d    = (const float*)d_in[7];
  const float* b_d    = (const float*)d_in[8];
  const float* W_attn = (const float*)d_in[9];
  const float* b_attn = (const float*)d_in[10];
  const float* W_1    = (const float*)d_in[11];
  const float* b_1    = (const float*)d_in[12];
  const float* W_2    = (const float*)d_in[13];
  const float* b_2    = (const float*)d_in[14];
  const float* W_3    = (const float*)d_in[15];
  const float* b_3    = (const float*)d_in[16];
  float* out = (float*)d_out;

  float* buf0 = (float*)d_ws;
  float* buf1 = buf0 + NTOT;
  float* buf2 = buf1 + NTOT;
  unsigned short* wb = (unsigned short*)(buf2 + NTOT);
  unsigned short* wb_attn = wb + 0 * 65536;
  unsigned short* wb_h    = wb + 1 * 65536;
  unsigned short* wb_w    = wb + 2 * 65536;
  unsigned short* wb_c    = wb + 3 * 65536;
  unsigned short* wb_1    = wb + 4 * 65536;
  unsigned short* wb_d    = wb + 5 * 65536;
  unsigned short* wb_2    = wb + 6 * 65536;
  unsigned short* wb_3    = wb + 7 * 65536;

  const dim3 blk(256), swgrid(128);
  swizzle_w_kernel<<<swgrid, blk, 0, stream>>>(W_attn, wb_attn);
  swizzle_w_kernel<<<swgrid, blk, 0, stream>>>(W_h,    wb_h);
  swizzle_w_kernel<<<swgrid, blk, 0, stream>>>(W_w,    wb_w);
  swizzle_w_kernel<<<swgrid, blk, 0, stream>>>(W_c,    wb_c);
  swizzle_w_kernel<<<swgrid, blk, 0, stream>>>(W_1,    wb_1);
  swizzle_w_kernel<<<swgrid, blk, 0, stream>>>(W_d,    wb_d);
  swizzle_w_kernel<<<swgrid, blk, 0, stream>>>(W_2,    wb_2);
  swizzle_w_kernel<<<swgrid, blk, 0, stream>>>(W_3,    wb_3);

  const dim3 ggrid(NROWS / 64);       // 2048 blocks per GEMM stage

  // ip_mlp: att -> buf0 ; (c, +h, +w) -> buf1 ; x1 = x + W1*((1+att)*sum) -> buf2
  gemm_stage_kernel<MODE_ATT,  EPI_STORE, false><<<ggrid, blk, 0, stream>>>(x, nullptr, wb_attn, b_attn, nullptr, buf0);
  gemm_stage_kernel<MODE_CONT, EPI_STORE, false><<<ggrid, blk, 0, stream>>>(x, nullptr, wb_c, b_c, nullptr, buf1);
  gemm_stage_kernel<MODE_H,    EPI_ACC,   false><<<ggrid, blk, 0, stream>>>(x, nullptr, wb_h, b_h, nullptr, buf1);
  gemm_stage_kernel<MODE_W,    EPI_ACC,   false><<<ggrid, blk, 0, stream>>>(x, nullptr, wb_w, b_w, nullptr, buf1);
  gemm_stage_kernel<MODE_CONT, EPI_RES,   true ><<<ggrid, blk, 0, stream>>>(buf0, buf1, wb_1, b_1, x, buf2);
  // tp_mlp: t_d -> buf0 ; x2 = x1 + W2*t_d -> buf1
  gemm_stage_kernel<MODE_D,    EPI_STORE, false><<<ggrid, blk, 0, stream>>>(buf2, nullptr, wb_d, b_d, nullptr, buf0);
  gemm_stage_kernel<MODE_CONT, EPI_RES,   false><<<ggrid, blk, 0, stream>>>(buf0, nullptr, wb_2, b_2, buf2, buf1);
  // final: out = x2 + W3*x2
  gemm_stage_kernel<MODE_CONT, EPI_RES,   false><<<ggrid, blk, 0, stream>>>(buf1, nullptr, wb_3, b_3, buf1, out);
}
